// LSTMComplex_51505247814030
// MI455X (gfx1250) — matrix operand order repaired and checked by reading, not compiled, not run on hardware
//
#include <hip/hip_runtime.h>
#include <hip/hip_bf16.h>

typedef __attribute__((ext_vector_type(16))) _Float16 v16h;
typedef __attribute__((ext_vector_type(8)))  float    v8f;

#define LN_EPS 1e-6f

union ABFrag { v16h v; unsigned u[8]; };
union HPack  { _Float16 h[2]; unsigned u; };

__device__ __forceinline__ float sigmoidf_(float x) { return 1.0f / (1.0f + __expf(-x)); }
// tanh via single fast exp: tanh(x) = 1 - 2/(exp(2x)+1). Saturates correctly at +-1.
__device__ __forceinline__ float tanhf_(float x)
{
    float e = __expf(2.0f * x);
    return 1.0f - 2.0f / (e + 1.0f);
}

// ---------------------------------------------------------------------------
// Grid-wide barrier (generation counter). All participating workgroups are
// co-resident by construction (<=16 WGs).
// ---------------------------------------------------------------------------
__device__ __forceinline__ void grid_barrier(unsigned* cnt, unsigned* gen, unsigned nwg)
{
    __threadfence();
    __syncthreads();
    if (threadIdx.x == 0) {
        unsigned g = __hip_atomic_load(gen, __ATOMIC_RELAXED, __HIP_MEMORY_SCOPE_AGENT);
        unsigned arrived = __hip_atomic_fetch_add(cnt, 1u, __ATOMIC_ACQ_REL, __HIP_MEMORY_SCOPE_AGENT);
        if (arrived == nwg - 1u) {
            __hip_atomic_store(cnt, 0u, __ATOMIC_RELAXED, __HIP_MEMORY_SCOPE_AGENT);
            __hip_atomic_fetch_add(gen, 1u, __ATOMIC_RELEASE, __HIP_MEMORY_SCOPE_AGENT);
        } else {
            while (__hip_atomic_load(gen, __ATOMIC_ACQUIRE, __HIP_MEMORY_SCOPE_AGENT) == g) {
                __builtin_amdgcn_s_sleep(1);
            }
        }
    }
    __syncthreads();
}

// ---------------------------------------------------------------------------
// B-fragment fetch from LDS (N-major staged weights) + one WMMA.
// B 32x16 f16 layout: dword v holds K pair at k = 2v + (v>=4 ? 8 : 0) +
// (lane>=16 ? 8 : 0); lane's column N = lane&15.
// ---------------------------------------------------------------------------
__device__ __forceinline__ v8f gate_mma(const _Float16* smB, int K, int nn, int hi,
                                        int kc, int g, v16h a, v8f acc)
{
    ABFrag bf;
    const _Float16* bp = smB + (size_t)(g * 16 + nn) * K + kc * 32 + hi * 8;
#pragma unroll
    for (int v = 0; v < 8; ++v)
        bf.u[v] = *reinterpret_cast<const unsigned*>(bp + 2 * v + ((v >= 4) ? 8 : 0));
    return __builtin_amdgcn_wmma_f32_16x16x32_f16(false, a, false, bf.v,
                                                  (short)0, acc, false, false);
}

// ---------------------------------------------------------------------------
// Persistent LSTM layer kernel.
//   xin  : [B=128][T][F] f32  (layer input sequence, already LayerNorm'd)
//   WxT  : [4H][F] f16 (transposed input weights)
//   WhT  : [4H][H] f16 (transposed recurrent weights)
//   bias : [4H] f32
//   hbuf : 2*[128][H] f16 ping-pong recurrent state
//   out  : SEQOUT ? [B][T][H] f32 : [B][H] f32 (raw h, LN applied later)
// Grid: H/16 workgroups x 256 threads. Each wave owns a 16-row batch tile;
// each WG owns a 16-column slice of H (same slice in all four gates).
// ---------------------------------------------------------------------------
template<int F, int H, bool SEQOUT>
__global__ __launch_bounds__(256)
void lstm_recur(const float* __restrict__ xin,
                const _Float16* __restrict__ WxT,
                const _Float16* __restrict__ WhT,
                const float* __restrict__ bias,
                _Float16* __restrict__ hbuf,
                float* __restrict__ out,
                unsigned* __restrict__ cnt, unsigned* __restrict__ gen,
                int T)
{
    constexpr int Bb = 128;
    constexpr int K  = F + H;          // concat [x_t | h_{t-1}] contraction dim
    constexpr int KC = K / 32;
    constexpr unsigned NWG = H / 16;

    __shared__ _Float16 smB[64 * K];   // 4 gate tiles x 16 cols, N-major over K

    const int wg   = blockIdx.x;
    const int wave = threadIdx.x >> 5;
    const int lane = threadIdx.x & 31;
    const int nn   = lane & 15;
    const int hi   = lane >> 4;

    // Stage [Wx;Wh] column slice for this WG into LDS (f16, N-major).
    for (int idx = threadIdx.x; idx < 64 * K; idx += 256) {
        int col  = idx / K;
        int k    = idx - col * K;
        int gate = col >> 4, c = col & 15;
        int gcol = gate * H + wg * 16 + c;
        smB[idx] = (k < F) ? WxT[(size_t)gcol * F + k]
                           : WhT[(size_t)gcol * H + (k - F)];
    }

    _Float16* hb0 = hbuf;
    _Float16* hb1 = hbuf + (size_t)Bb * H;
    // h_{-1} = 0: each WG zeroes its own column slice of the ping buffer.
    for (int idx = threadIdx.x; idx < Bb * 16; idx += 256) {
        int b = idx >> 4, c = idx & 15;
        hb0[(size_t)b * H + wg * 16 + c] = (_Float16)0.f;
    }
    __syncthreads();
    grid_barrier(cnt, gen, NWG);

    const int colH = wg * 16 + nn;
    const float bi0 = bias[0 * H + colH];
    const float bi1 = bias[1 * H + colH];
    const float bi2 = bias[2 * H + colH];
    const float bi3 = bias[3 * H + colH];

    const int brow = wave * 16 + nn;   // A-fragment batch row (lane&15)
    v8f cc = {};                       // persistent cell-state tile

    for (int t = 0; t < T; ++t) {
        const _Float16* hread = (t & 1) ? hb1 : hb0;
        _Float16* hwrite      = (t & 1) ? hb0 : hb1;

        v8f acc0 = {}, acc1 = {}, acc2 = {}, acc3 = {};
        const float* xrow = xin + ((size_t)brow * T + t) * F;

        // Pull next timestep's input row toward the caches while this step's
        // WMMA chain and the inter-WG barrier run (global_prefetch_b8).
        if (t + 1 < T) __builtin_prefetch(xrow + F, 0, 0);

#pragma unroll
        for (int kc = 0; kc < KC; ++kc) {
            // A 16x32 f16 layout: row = lane&15; dword v holds K pair at
            // k = kc*32 + 2v + (v>=4 ? 8 : 0) + (lane>=16 ? 8 : 0).
            ABFrag a;
            if (kc * 32 < F) {         // chunk from x_t (f32 -> f16 convert)
#pragma unroll
                for (int v = 0; v < 8; ++v) {
                    int k = kc * 32 + 2 * v + ((v >= 4) ? 8 : 0) + hi * 8;
                    float2 f2 = *reinterpret_cast<const float2*>(xrow + k);
                    HPack pk; pk.h[0] = (_Float16)f2.x; pk.h[1] = (_Float16)f2.y;
                    a.u[v] = pk.u;
                }
            } else {                    // chunk from h_{t-1} (already f16)
                const _Float16* hrow = hread + (size_t)brow * H - F;
#pragma unroll
                for (int v = 0; v < 8; ++v) {
                    int k = kc * 32 + 2 * v + ((v >= 4) ? 8 : 0) + hi * 8;
                    a.u[v] = *reinterpret_cast<const unsigned*>(hrow + k);
                }
            }
            acc0 = gate_mma(smB, K, nn, hi, kc, 0, a.v, acc0);
            acc1 = gate_mma(smB, K, nn, hi, kc, 1, a.v, acc1);
            acc2 = gate_mma(smB, K, nn, hi, kc, 2, a.v, acc2);
            acc3 = gate_mma(smB, K, nn, hi, kc, 3, a.v, acc3);
        }

        // Gates + state update. C/D layout: slot r -> row = r + 8*(lane>=16),
        // col = lane&15.
        v8f hh;
#pragma unroll
        for (int r = 0; r < 8; ++r) {
            float iv = sigmoidf_(acc0[r] + bi0);
            float fv = sigmoidf_(acc1[r] + bi1);
            float gv = tanhf_(acc2[r] + bi2);
            float ov = sigmoidf_(acc3[r] + bi3);
            float cv = fv * cc[r] + iv * gv;
            cc[r] = cv;
            float hv = ov * tanhf_(cv);
            hh[r] = hv;
            int b2 = wave * 16 + hi * 8 + r;
            // Recurrent f16 state: must be visible to all WGs before barrier.
            hwrite[(size_t)b2 * H + colH] = (_Float16)hv;
            if (!SEQOUT && t == T - 1) {
                out[(size_t)b2 * H + colH] = hv;
            }
        }
        grid_barrier(cnt, gen, NWG);

        // Raw sequence output (consumed only by later stream-ordered kernels):
        // issued after the barrier so its store latency never gates the fence.
        if (SEQOUT) {
#pragma unroll
            for (int r = 0; r < 8; ++r) {
                int b2 = wave * 16 + hi * 8 + r;
                out[((size_t)b2 * T + t) * H + colH] = hh[r];
            }
        }
    }
}

// ---------------------------------------------------------------------------
// In-place row LayerNorm, one wave (32 lanes) per row of length H.
// ---------------------------------------------------------------------------
template<int H>
__global__ __launch_bounds__(256)
void ln_rows_inplace(float* __restrict__ data, const float* __restrict__ sc,
                     const float* __restrict__ bi, int rows)
{
    constexpr int NP = H / 32;
    int wave = threadIdx.x >> 5, lane = threadIdx.x & 31;
    int row = blockIdx.x * 8 + wave;
    if (row >= rows) return;
    float* p = data + (size_t)row * H;
    float v[NP];
    float s = 0.f;
#pragma unroll
    for (int j = 0; j < NP; ++j) { v[j] = p[lane + j * 32]; s += v[j]; }
#pragma unroll
    for (int o = 16; o > 0; o >>= 1) s += __shfl_xor(s, o, 32);
    float m = s / (float)H;
    float q = 0.f;
#pragma unroll
    for (int j = 0; j < NP; ++j) { float d = v[j] - m; q += d * d; }
#pragma unroll
    for (int o = 16; o > 0; o >>= 1) q += __shfl_xor(q, o, 32);
    float rs = rsqrtf(q / (float)H + LN_EPS);
#pragma unroll
    for (int j = 0; j < NP; ++j) {
        int c = lane + j * 32;
        p[c] = (v[j] - m) * rs * sc[c] + bi[c];
    }
}

// ---------------------------------------------------------------------------
// Weight prep: f32 [R][C] -> f16 [C][R] (transposed, B-fragment friendly).
// ---------------------------------------------------------------------------
__global__ __launch_bounds__(256)
void transpose_f32_f16(const float* __restrict__ in, _Float16* __restrict__ out,
                       int R, int C)
{
    int idx = blockIdx.x * 256 + threadIdx.x;
    if (idx < R * C) {
        int r = idx / C, c = idx - r * C;
        out[(size_t)c * R + r] = (_Float16)in[(size_t)r * C + c];
    }
}

__global__ void init_sync(unsigned* s)
{
    if (threadIdx.x < 16) s[threadIdx.x] = 0u;
}

// ---------------------------------------------------------------------------
// MLP head: LN3 -> (Dense 64->128, LN, ReLU) -> (128->64, LN, ReLU)
//           -> (64->32, LN, ReLU) -> (32->1) -> ReLU.   One block per row.
// ---------------------------------------------------------------------------
__device__ void blk_ln(float* a, int n, const float* sc, const float* bi,
                       float* red, bool do_relu)
{
    __syncthreads();
    if (threadIdx.x == 0) {
        float s = 0.f;
        for (int i = 0; i < n; ++i) s += a[i];
        float m = s / (float)n;
        float q = 0.f;
        for (int i = 0; i < n; ++i) { float d = a[i] - m; q += d * d; }
        red[0] = m;
        red[1] = rsqrtf(q / (float)n + LN_EPS);
    }
    __syncthreads();
    if ((int)threadIdx.x < n) {
        float x = (a[threadIdx.x] - red[0]) * red[1] * sc[threadIdx.x] + bi[threadIdx.x];
        if (do_relu) x = fmaxf(x, 0.f);
        a[threadIdx.x] = x;
    }
    __syncthreads();
}

__global__ __launch_bounds__(128)
void head_kernel(const float* __restrict__ h3,
                 const float* s3, const float* b3,
                 const float* Wd1, const float* bd1, const float* s1, const float* e1,
                 const float* Wd2, const float* bd2, const float* s2, const float* e2,
                 const float* Wd3, const float* bd3, const float* s3h, const float* e3,
                 const float* Wout, const float* bout, float* __restrict__ outp)
{
    int b = blockIdx.x, tid = threadIdx.x;
    __shared__ float a0[64], a1[128], a2[64], a3[32];
    __shared__ float red[2];

    if (tid < 64) a0[tid] = h3[(size_t)b * 64 + tid];
    blk_ln(a0, 64, s3, b3, red, false);

    if (tid < 128) {
        float acc = bd1[tid];
        for (int k = 0; k < 64; ++k) acc += a0[k] * Wd1[k * 128 + tid];
        a1[tid] = acc;
    }
    blk_ln(a1, 128, s1, e1, red, true);

    if (tid < 64) {
        float acc = bd2[tid];
        for (int k = 0; k < 128; ++k) acc += a1[k] * Wd2[k * 64 + tid];
        a2[tid] = acc;
    }
    blk_ln(a2, 64, s2, e2, red, true);

    if (tid < 32) {
        float acc = bd3[tid];
        for (int k = 0; k < 64; ++k) acc += a2[k] * Wd3[k * 32 + tid];
        a3[tid] = acc;
    }
    blk_ln(a3, 32, s3h, e3, red, true);

    if (tid == 0) {
        float acc = bout[0];
        for (int k = 0; k < 32; ++k) acc += a3[k] * Wout[k];
        outp[b] = fmaxf(acc, 0.f);
    }
}

// ---------------------------------------------------------------------------
extern "C" void kernel_launch(void* const* d_in, const int* in_sizes, int n_in,
                              void* d_out, int out_size, void* d_ws, size_t ws_size,
                              hipStream_t stream)
{
    (void)in_sizes; (void)n_in; (void)out_size; (void)ws_size;

    const float* x    = (const float*)d_in[0];
    const float* Wx1  = (const float*)d_in[1];
    const float* Wh1  = (const float*)d_in[2];
    const float* b1   = (const float*)d_in[3];
    const float* ln1s = (const float*)d_in[4];
    const float* ln1b = (const float*)d_in[5];
    const float* Wx2  = (const float*)d_in[6];
    const float* Wh2  = (const float*)d_in[7];
    const float* b2   = (const float*)d_in[8];
    const float* ln2s = (const float*)d_in[9];
    const float* ln2b = (const float*)d_in[10];
    const float* Wx3  = (const float*)d_in[11];
    const float* Wh3  = (const float*)d_in[12];
    const float* b3   = (const float*)d_in[13];
    const float* ln3s = (const float*)d_in[14];
    const float* ln3b = (const float*)d_in[15];
    const float* Wd1  = (const float*)d_in[16];
    const float* bd1  = (const float*)d_in[17];
    const float* l1s  = (const float*)d_in[18];
    const float* l1b  = (const float*)d_in[19];
    const float* Wd2  = (const float*)d_in[20];
    const float* bd2  = (const float*)d_in[21];
    const float* l2s  = (const float*)d_in[22];
    const float* l2b  = (const float*)d_in[23];
    const float* Wd3  = (const float*)d_in[24];
    const float* bd3  = (const float*)d_in[25];
    const float* l3s  = (const float*)d_in[26];
    const float* l3b  = (const float*)d_in[27];
    const float* Wout = (const float*)d_in[28];
    const float* bout = (const float*)d_in[29];
    float* outp = (float*)d_out;

    // Workspace carve-up (256B aligned).
    char* p = (char*)d_ws;
    auto take = [&](size_t bytes) -> void* {
        void* r = (void*)p;
        p += (bytes + 255) & ~(size_t)255;
        return r;
    };
    unsigned*  sync  = (unsigned*)take(64 * sizeof(unsigned));
    _Float16*  WxT1  = (_Float16*)take((size_t)1024 * 64  * 2);
    _Float16*  WhT1  = (_Float16*)take((size_t)1024 * 256 * 2);
    _Float16*  WxT2  = (_Float16*)take((size_t)512  * 256 * 2);
    _Float16*  WhT2  = (_Float16*)take((size_t)512  * 128 * 2);
    _Float16*  WxT3  = (_Float16*)take((size_t)256  * 128 * 2);
    _Float16*  WhT3  = (_Float16*)take((size_t)256  * 64  * 2);
    _Float16*  hbuf1 = (_Float16*)take((size_t)2 * 128 * 256 * 2);
    _Float16*  hbuf2 = (_Float16*)take((size_t)2 * 128 * 128 * 2);
    _Float16*  hbuf3 = (_Float16*)take((size_t)2 * 128 * 64  * 2);
    float*     hseq1 = (float*)take((size_t)128 * 1024 * 256 * 4);
    float*     hseq2 = (float*)take((size_t)128 * 1024 * 128 * 4);
    float*     h3raw = (float*)take((size_t)128 * 64 * 4);

    init_sync<<<1, 64, 0, stream>>>(sync);

    auto tr = [&](const float* in, _Float16* out, int R, int C) {
        int n = R * C;
        transpose_f32_f16<<<(n + 255) / 256, 256, 0, stream>>>(in, out, R, C);
    };
    tr(Wx1, WxT1, 64,  1024);
    tr(Wh1, WhT1, 256, 1024);
    tr(Wx2, WxT2, 256, 512);
    tr(Wh2, WhT2, 128, 512);
    tr(Wx3, WxT3, 128, 256);
    tr(Wh3, WhT3, 64,  256);

    const int T = 1024;
    const int rows = 128 * 1024;

    lstm_recur<64, 256, true><<<16, 256, 0, stream>>>(
        x, WxT1, WhT1, b1, hbuf1, hseq1, sync + 0, sync + 1, T);
    ln_rows_inplace<256><<<rows / 8, 256, 0, stream>>>(hseq1, ln1s, ln1b, rows);

    lstm_recur<256, 128, true><<<8, 256, 0, stream>>>(
        hseq1, WxT2, WhT2, b2, hbuf2, hseq2, sync + 2, sync + 3, T);
    ln_rows_inplace<128><<<rows / 8, 256, 0, stream>>>(hseq2, ln2s, ln2b, rows);

    lstm_recur<128, 64, false><<<4, 256, 0, stream>>>(
        hseq2, WxT3, WhT3, b3, hbuf3, h3raw, sync + 4, sync + 5, T);

    head_kernel<<<128, 128, 0, stream>>>(
        h3raw, ln3s, ln3b, Wd1, bd1, l1s, l1b, Wd2, bd2, l2s, l2b,
        Wd3, bd3, l3s, l3b, Wout, bout, outp);
}
